// ViG_Model_18769007083724
// MI455X (gfx1250) — matrix operand order repaired and checked by reading, not compiled, not run on hardware
//
#include <hip/hip_runtime.h>

// ---------------------------------------------------------------- types
typedef __attribute__((ext_vector_type(16))) __bf16        v16bf;
typedef __attribute__((ext_vector_type(8)))  float         v8f;
typedef __attribute__((ext_vector_type(4))) unsigned int   uint4v;
typedef int v4i __attribute__((vector_size(16)));          // matches builtin param type
typedef long long ll;

#define BM 64
#define BN 128
#define BK 32
#define KP (BK / 2)   // K-pairs (one packed bf16x2 dword per pair)

#define AS1 __attribute__((address_space(1)))
#define AS3 __attribute__((address_space(3)))

#if defined(__has_builtin)
#if __has_builtin(__builtin_amdgcn_global_load_async_to_lds_b128) && \
    __has_builtin(__builtin_amdgcn_s_wait_asynccnt)
#define HAVE_ASYNC_LDS 1
#endif
#endif

__device__ __forceinline__ float gelu_exact(float x) {
  return 0.5f * x * (1.0f + erff(x * 0.70710678118654752f));
}

__device__ __forceinline__ unsigned short f2bf(float f) {
  unsigned int u = __float_as_uint(f);
  u += 0x7FFFu + ((u >> 16) & 1u);          // round-to-nearest-even
  return (unsigned short)(u >> 16);
}

__device__ __forceinline__ unsigned pack_bf16x2(float lo, float hi) {
  return (unsigned)f2bf(lo) | ((unsigned)f2bf(hi) << 16);
}

// ---------------------------------------------------------------- WMMA GEMM
// C[b] = A (M x K)  x  B[b] (K x N), bf16 inputs (converted during LDS staging),
// f32 accumulate via v_wmma_f32_16x16x32_bf16. Block tile 64x128, 4 waves,
// each wave: 32x64 = 2x4 accumulators -> 8 WMMA per K-step.
// epi == 0 : y = acc*scale[m] + (bias[m]*scale[m] + offs[m])   (folded conv-bias + BN)
// epi == 1 : y = 2*acc - sq[m] - sq[n] + relpos[m*N+n]         (grapher score)
// epi == 2 : y = acc
// then optional exact GELU and optional residual add.
__global__ __launch_bounds__(128)
void wmma_gemm_kernel(const float* __restrict__ A, const float* __restrict__ B,
                      float* __restrict__ C,
                      int M, int N, int K,
                      ll sA, ll sB, ll sC,
                      int lda, int ldb, int ldc, int a_trans,
                      int epi,
                      const float* __restrict__ bias,
                      const float* __restrict__ scale,
                      const float* __restrict__ offs,
                      int act,
                      const float* __restrict__ res, ll sRes,
                      const float* __restrict__ sq, ll sSq,
                      const float* __restrict__ relpos) {
  __shared__ alignas(16) unsigned lA32[BM * KP];   // [m][kpair], packed bf16x2
  __shared__ alignas(16) unsigned lB32[BN * KP];   // [n][kpair], packed bf16x2

  const int tid  = threadIdx.x;
  const int wave = tid >> 5;
  const int lane = tid & 31;
  const int half = lane >> 4;
  const int l16  = lane & 15;
  const int wm   = (wave & 1) * 32;
  const int wn   = (wave >> 1) * 64;
  const int bm0  = blockIdx.y * BM;
  const int bn0  = blockIdx.x * BN;
  const int bat  = blockIdx.z;

  const float* Ab = A + (ll)bat * sA;
  const float* Bb = B + (ll)bat * sB;
  float*       Cb = C + (ll)bat * sC;

  v8f acc[2][4] = {};

  for (int k0 = 0; k0 < K; k0 += BK) {
    __syncthreads();
    // ---- stage A tile: pack 2 k-values per dword ----
    for (int i = tid; i < BM * KP; i += 128) {
      int r, c2;
      if (a_trans) { r = i & (BM - 1); c2 = i >> 6; }   // coalesce over m in A[k*lda+m]
      else         { r = i >> 4;       c2 = i & (KP - 1); } // near-coalesced float2 over k
      int gm = bm0 + r;
      int gk0 = k0 + 2 * c2, gk1 = gk0 + 1;
      float v0 = 0.0f, v1 = 0.0f;
      if (gm < M) {
        if (a_trans) {
          if (gk0 < K) v0 = Ab[(ll)gk0 * lda + gm];
          if (gk1 < K) v1 = Ab[(ll)gk1 * lda + gm];
        } else {
          if (gk0 < K) v0 = Ab[(ll)gm * lda + gk0];
          if (gk1 < K) v1 = Ab[(ll)gm * lda + gk1];
        }
      }
      lA32[r * KP + c2] = pack_bf16x2(v0, v1);
    }
    // ---- stage B tile transposed into [n][kpair] (coalesced over n) ----
    for (int i = tid; i < BN * KP; i += 128) {
      int n = i & (BN - 1), c2 = i >> 7;
      int gn = bn0 + n;
      int gk0 = k0 + 2 * c2, gk1 = gk0 + 1;
      float v0 = 0.0f, v1 = 0.0f;
      if (gn < N) {
        if (gk0 < K) v0 = Bb[(ll)gk0 * ldb + gn];
        if (gk1 < K) v1 = Bb[(ll)gk1 * ldb + gn];
      }
      lB32[n * KP + c2] = pack_bf16x2(v0, v1);
    }
    // ---- prefetch next K tile of B (global_prefetch_b8) ----
    if (k0 + BK < K) {
      int gn = bn0 + tid;
      int gk = k0 + BK;
      if (gn < N) __builtin_prefetch(&Bb[(ll)gk * ldb + gn], 0, 1);
    }
    __syncthreads();

    // ---- build fragments per ISA 16-bit A(16x32) / B(32x16) layouts ----
    v16bf fa[2], fb[4];
#pragma unroll
    for (int t = 0; t < 2; ++t) {
      const unsigned* pa = &lA32[(wm + t * 16 + l16) * KP];
      union { uint4v u[2]; v16bf v; } fu;
      fu.u[0] = *reinterpret_cast<const uint4v*>(pa + half * 4);       // K = h*8 .. h*8+7
      fu.u[1] = *reinterpret_cast<const uint4v*>(pa + half * 4 + 8);   // K = 16+h*8 ..
      fa[t] = fu.v;
    }
#pragma unroll
    for (int t = 0; t < 4; ++t) {
      const unsigned* pb = &lB32[(wn + t * 16 + l16) * KP];
      union { uint4v u[2]; v16bf v; } gu;
      gu.u[0] = *reinterpret_cast<const uint4v*>(pb + half * 4);       // K = h*8 .. h*8+7
      gu.u[1] = *reinterpret_cast<const uint4v*>(pb + half * 4 + 8);   // K = 16+h*8 ..
      fb[t] = gu.v;
    }
#pragma unroll
    for (int tm = 0; tm < 2; ++tm)
#pragma unroll
      for (int tn = 0; tn < 4; ++tn)
        acc[tm][tn] = __builtin_amdgcn_wmma_f32_16x16x32_bf16(
            false, fa[tm], false, fb[tn], (short)0, acc[tm][tn], false, false);
  }

  const float* resb = res ? res + (ll)bat * sRes : nullptr;
  const float* sqb  = sq  ? sq  + (ll)bat * sSq  : nullptr;

#pragma unroll
  for (int tm = 0; tm < 2; ++tm)
    for (int tn = 0; tn < 4; ++tn)
#pragma unroll
      for (int r = 0; r < 8; ++r) {
        int gm = bm0 + wm + tm * 16 + half * 8 + r;  // C/D layout: VGPR r -> M=r (+8 hi half)
        int gn = bn0 + wn + tn * 16 + l16;
        if (gm < M && gn < N) {
          float v = acc[tm][tn][r];
          if (epi == 0) {
            float sc = scale[gm];
            v = v * sc + (bias[gm] * sc + offs[gm]);
          } else if (epi == 1) {
            v = 2.0f * v - sqb[gm] - sqb[gn] + relpos[(ll)gm * N + gn];
          }
          if (act) v = gelu_exact(v);
          if (resb) v += resb[(ll)gm * ldc + gn];
          Cb[(ll)gm * ldc + gn] = v;
        }
      }
}

// ---------------------------------------------------------------- direct 3x3 conv + BN (+GELU)
__global__ void conv3x3_bn_kernel(const float* __restrict__ in, const float* __restrict__ w,
                                  const float* __restrict__ bias, const float* __restrict__ s,
                                  const float* __restrict__ o, float* __restrict__ out,
                                  int Bt, int Cin, int Hin, int Win,
                                  int Cout, int stride, int Hout, int Wout, int act) {
  ll idx = (ll)blockIdx.x * blockDim.x + threadIdx.x;
  ll total = (ll)Bt * Cout * Hout * Wout;
  if (idx >= total) return;
  int x = idx % Wout; ll t = idx / Wout;
  int y = t % Hout;   t /= Hout;
  int co = t % Cout;  int b = t / Cout;
  float acc = bias[co];
  for (int ci = 0; ci < Cin; ++ci) {
    const float* ip = in + ((ll)(b * Cin + ci)) * Hin * Win;
    const float* wp = w + ((ll)(co * Cin + ci)) * 9;
    for (int ky = 0; ky < 3; ++ky) {
      int iy = y * stride + ky - 1;
      if (iy < 0 || iy >= Hin) continue;
      for (int kx = 0; kx < 3; ++kx) {
        int ix = x * stride + kx - 1;
        if (ix < 0 || ix >= Win) continue;
        acc += ip[(ll)iy * Win + ix] * wp[ky * 3 + kx];
      }
    }
  }
  acc = acc * s[co] + o[co];
  if (act) acc = gelu_exact(acc);
  out[idx] = acc;
}

// ---------------------------------------------------------------- pos-embed add
__global__ void addpos_kernel(float* __restrict__ h, const float* __restrict__ pe,
                              ll per_batch, int Bt) {
  ll i = (ll)blockIdx.x * blockDim.x + threadIdx.x;
  if (i >= per_batch * Bt) return;
  h[i] += pe[i % per_batch];
}

// ---------------------------------------------------------------- column L2 norm
__global__ void colnorm_kernel(const float* __restrict__ feat, float* __restrict__ xn,
                               float* __restrict__ sq, int C, int N) {
  int b = blockIdx.y;
  int n = blockIdx.x * 256 + threadIdx.x;
  if (n >= N) return;
  const float* f = feat + (ll)b * C * N;
  float s = 0.0f;
  for (int c = 0; c < C; ++c) { float v = f[(ll)c * N + n]; s += v * v; }
  float inv = 1.0f / (sqrtf(s) + 1e-12f);
  float* xo = xn + (ll)b * C * N;
  float s2 = 0.0f;
  for (int c = 0; c < C; ++c) {
    float v = f[(ll)c * N + n] * inv;
    xo[(ll)c * N + n] = v;
    s2 += v * v;
  }
  sq[b * N + n] = s2;
}

// ---------------------------------------------------------------- (C,N) -> (N,C) transpose
__global__ void transpose_kernel(const float* __restrict__ in, float* __restrict__ out,
                                 int C, int N, int Bt) {
  ll i = (ll)blockIdx.x * blockDim.x + threadIdx.x;
  ll total = (ll)Bt * C * N;
  if (i >= total) return;
  int n = i % N; ll r = i / N;
  int c = r % C; int b = (int)(r / C);
  out[((ll)b * N + n) * C + c] = in[i];
}

// ---------------------------------------------------------------- per-row top-k + msg + concat
// One wave32 per (b, m). Score row staged into LDS via async global->LDS copy
// (ASYNCcnt path) when 16B-aligned, then iterative wave-argmax selection.
// Writes concat[b] (2C x N): rows [0,C) = feat, rows [C,2C) = msg.
__global__ void topk_msg_kernel(const float* __restrict__ score, const float* __restrict__ featT,
                                float* __restrict__ concat, int N, int C, int kk, int dil) {
  extern __shared__ float smem[];
  int* sidx = (int*)(smem + N);
  int bm = blockIdx.x;
  int b = bm / N, m = bm % N;
  int lane = threadIdx.x;
  const float* row = score + (ll)bm * N;
#ifdef HAVE_ASYNC_LDS
  if ((N & 3) == 0) {
    for (int i = lane; i < (N >> 2); i += 32) {
      __builtin_amdgcn_global_load_async_to_lds_b128(
          (AS1 v4i*)(unsigned long long)(row + 4 * i),
          (AS3 v4i*)(unsigned)(unsigned long long)(smem + 4 * i),
          0, 0);
    }
    __builtin_amdgcn_s_wait_asynccnt(0);
  } else
#endif
  {
    for (int i = lane; i < N; i += 32) smem[i] = row[i];
  }
  __syncthreads();
  int kd = kk * dil;
  for (int t = 0; t < kd; ++t) {
    float bv = -3.0e38f; int bi = 0;
    for (int i = lane; i < N; i += 32) {
      float v = smem[i];
      if (v > bv) { bv = v; bi = i; }
    }
    for (int off = 16; off > 0; off >>= 1) {
      float ov = __shfl_down(bv, off, 32);
      int   oi = __shfl_down(bi, off, 32);
      if (ov > bv || (ov == bv && oi < bi)) { bv = ov; bi = oi; }  // stable: lower idx wins ties
    }
    bi = __shfl(bi, 0, 32);
    if (lane == 0) { sidx[t] = bi; smem[bi] = -3.0e38f; }
    __syncthreads();
  }
  const float* fT = featT + (ll)b * N * C;
  for (int c = lane; c < C; c += 32) {
    float self = fT[(ll)m * C + c];
    float best = -3.0e38f;
    for (int j = 0; j < 9; ++j) {
      int nb = sidx[j * dil];
      best = fmaxf(best, fT[(ll)nb * C + c]);
    }
    concat[((ll)b * 2 * C + c) * N + m]     = self;         // feat half
    concat[((ll)b * 2 * C + C + c) * N + m] = best - self;  // msg half
  }
}

// ---------------------------------------------------------------- global average pool
__global__ void avgpool_kernel(const float* __restrict__ in, float* __restrict__ out,
                               int total_bc, int N) {
  int i = blockIdx.x * 256 + threadIdx.x;
  if (i >= total_bc) return;
  const float* p = in + (ll)i * N;
  float s = 0.0f;
  for (int n = 0; n < N; ++n) s += p[n];
  out[i] = s / (float)N;
}

// ---------------------------------------------------------------- head matvec
__global__ void matvec_kernel(const float* __restrict__ in, const float* __restrict__ w,
                              const float* __restrict__ bias, const float* __restrict__ s,
                              const float* __restrict__ o, float* __restrict__ out,
                              int Bt, int M, int K, int bn, int act) {
  int i = blockIdx.x * 256 + threadIdx.x;
  if (i >= Bt * M) return;
  int m = i % M, b = i / M;
  const float* x  = in + (ll)b * K;
  const float* wr = w + (ll)m * K;
  float acc = 0.0f;
  for (int k = 0; k < K; ++k) acc += wr[k] * x[k];
  acc += bias[m];
  if (bn)  acc = acc * s[m] + o[m];
  if (act) acc = gelu_exact(acc);
  out[i] = acc;
}

// ================================================================ host side
struct CBN { const float *w, *b, *s, *o; };

static void gemm(hipStream_t st, const float* A, const float* B, float* C,
                 int M, int N, int K, ll sA, ll sB, ll sC,
                 int lda, int ldb, int ldc, int a_trans, int epi,
                 const float* bias, const float* scale, const float* offs,
                 int act, const float* res, ll sRes,
                 const float* sq, ll sSq, const float* relpos, int Bt) {
  dim3 g((N + BN - 1) / BN, (M + BM - 1) / BM, Bt);
  wmma_gemm_kernel<<<g, dim3(128), 0, st>>>(A, B, C, M, N, K, sA, sB, sC,
                                            lda, ldb, ldc, a_trans, epi,
                                            bias, scale, offs, act, res, sRes,
                                            sq, sSq, relpos);
}

static void cbn_gemm(hipStream_t st, const CBN& p, const float* X, float* Y,
                     int M, int K, int Nsp, int act, const float* res, int Bt) {
  gemm(st, p.w, X, Y, M, Nsp, K, 0, (ll)K * Nsp, (ll)M * Nsp,
       K, Nsp, Nsp, 0, 0, p.b, p.s, p.o, act, res, res ? (ll)M * Nsp : 0,
       nullptr, 0, nullptr, Bt);
}

static void conv3x3(hipStream_t st, const float* in, const CBN& p, float* out,
                    int Bt, int Cin, int Hin, int Win, int Cout, int stride, int act) {
  int Hout = (stride == 2) ? Hin / 2 : Hin;
  int Wout = (stride == 2) ? Win / 2 : Win;
  ll total = (ll)Bt * Cout * Hout * Wout;
  conv3x3_bn_kernel<<<(unsigned)((total + 255) / 256), 256, 0, st>>>(
      in, p.w, p.b, p.s, p.o, out, Bt, Cin, Hin, Win, Cout, stride, Hout, Wout, act);
}

extern "C" void kernel_launch(void* const* d_in, const int* in_sizes, int n_in,
                              void* d_out, int out_size, void* d_ws, size_t ws_size,
                              hipStream_t stream) {
  (void)in_sizes; (void)n_in; (void)out_size; (void)ws_size;
  const int Bt = 2;
  int p = 0;
  auto F = [&](int i) { return reinterpret_cast<const float*>(d_in[i]); };
  auto getcbn = [&]() { CBN c; c.w = F(p++); c.b = F(p++); c.s = F(p++); c.o = F(p++); return c; };

  // ---- unpack inputs in setup_inputs() insertion order ----
  const float* x = F(p++);
  CBN stem0 = getcbn(), stem1 = getcbn(), stem2 = getcbn();
  const float* pos = F(p++);
  CBN down[3]; for (int i = 0; i < 3; ++i) down[i] = getcbn();
  struct Graph { CBN fc1, gc, fc2; const float* rel; } gr[12];
  for (int i = 0; i < 12; ++i) { gr[i].fc1 = getcbn(); gr[i].gc = getcbn(); gr[i].fc2 = getcbn(); gr[i].rel = F(p++); }
  struct Ffn { CBN fc1, fc2; } ff[12];
  for (int i = 0; i < 12; ++i) { ff[i].fc1 = getcbn(); ff[i].fc2 = getcbn(); }
  CBN hc0 = getcbn();
  const float* w1 = F(p++);
  const float* b1 = F(p++);

  // ---- workspace layout (floats) ----
  float* ws = reinterpret_cast<float*>(d_ws);
  const ll A_MAX = 2LL * 192 * 3136;        // covers all activation tensors incl. 4C hidden
  float* hA     = ws;
  float* hB     = hA + A_MAX;
  float* t1     = hB + A_MAX;               // feat / ffn hidden
  float* t2     = t1 + A_MAX;               // concat (2C, N)
  float* t3     = t2 + A_MAX;               // gc output (2C, N)
  float* xnb    = t3 + A_MAX;               // normalized features
  float* ftT    = xnb + A_MAX;              // feat transposed (N, C)
  float* sqb    = ftT + A_MAX;              // (B, N)
  float* poolb  = sqb + 8192;               // (B, 384)
  float* h1b    = poolb + 4096;             // (B, 1024)
  float* scoreb = h1b + 4096;               // (B, N, N) max 2*3136*3136

  // ---- stem ----
  conv3x3(stream, x,  stem0, hA, Bt,  3, 224, 224, 24, 2, 1);
  conv3x3(stream, hA, stem1, hB, Bt, 24, 112, 112, 48, 2, 1);
  conv3x3(stream, hB, stem2, hA, Bt, 48,  56,  56, 48, 1, 0);
  {
    ll per = 48LL * 56 * 56;
    addpos_kernel<<<(unsigned)((per * Bt + 255) / 256), 256, 0, stream>>>(hA, pos, per, Bt);
  }
  float* h = hA; float* halt = hB;

  const int CFGc[4] = {48, 96, 240, 384};
  const int BLK[4]  = {2, 2, 6, 2};
  const int HWs[4]  = {56, 28, 14, 7};
  const int dil_all[12] = {1,1,1,1, 2,2,2,2, 3,3,3,3};

  int bidx = 0;
  for (int s = 0; s < 4; ++s) {
    const int C = CFGc[s], HW = HWs[s], Nsp = HW * HW;
    if (s > 0) {
      conv3x3(stream, h, down[s - 1], halt, Bt, CFGc[s - 1], HWs[s - 1], HWs[s - 1], C, 2, 0);
      float* t = h; h = halt; halt = t;
    }
    for (int blk = 0; blk < BLK[s]; ++blk, ++bidx) {
      const Graph& G = gr[bidx];
      const int dil = dil_all[bidx];
      // grapher: fc1 (C x C)
      cbn_gemm(stream, G.fc1, h, t1, C, C, Nsp, 0, nullptr, Bt);
      // column normalize
      colnorm_kernel<<<dim3((Nsp + 255) / 256, Bt), 256, 0, stream>>>(t1, xnb, sqb, C, Nsp);
      // score = -dist + rel_pos  via  2*xn^T xn - sq_m - sq_n + relpos   (A transposed)
      gemm(stream, xnb, xnb, scoreb, Nsp, Nsp, C,
           (ll)C * Nsp, (ll)C * Nsp, (ll)Nsp * Nsp,
           Nsp, Nsp, Nsp, 1, 1, nullptr, nullptr, nullptr,
           0, nullptr, 0, sqb, Nsp, G.rel, Bt);
      // feat transpose for coalesced gathers
      {
        ll tot = (ll)Bt * C * Nsp;
        transpose_kernel<<<(unsigned)((tot + 255) / 256), 256, 0, stream>>>(t1, ftT, C, Nsp, Bt);
      }
      // top-k + max aggregation + concat build
      size_t shmem = (size_t)(Nsp + 64) * sizeof(float);
      topk_msg_kernel<<<Bt * Nsp, 32, shmem, stream>>>(scoreb, ftT, t2, Nsp, C, 9, dil);
      // gc (2C x 2C) + GELU
      cbn_gemm(stream, G.gc, t2, t3, 2 * C, 2 * C, Nsp, 1, nullptr, Bt);
      // fc2 (C x 2C) + residual
      cbn_gemm(stream, G.fc2, t3, h, C, 2 * C, Nsp, 0, h, Bt);
      // FFN
      const Ffn& Ff = ff[bidx];
      cbn_gemm(stream, Ff.fc1, h, t1, 4 * C, C, Nsp, 1, nullptr, Bt);
      cbn_gemm(stream, Ff.fc2, t1, h, C, 4 * C, Nsp, 0, h, Bt);
    }
  }

  // ---- head ----
  avgpool_kernel<<<(Bt * 384 + 255) / 256, 256, 0, stream>>>(h, poolb, Bt * 384, 49);
  matvec_kernel<<<(Bt * 1024 + 255) / 256, 256, 0, stream>>>(
      poolb, hc0.w, hc0.b, hc0.s, hc0.o, h1b, Bt, 1024, 384, 1, 1);
  matvec_kernel<<<(Bt * 1000 + 255) / 256, 256, 0, stream>>>(
      h1b, w1, b1, nullptr, nullptr, reinterpret_cast<float*>(d_out), Bt, 1000, 1024, 0, 0);
}
